// MultiHeadAttention_9826885173780
// MI455X (gfx1250) — compile-verified, matrix operand order checked by reading, not run on hardware
//
#include <hip/hip_runtime.h>
#include <hip/hip_bf16.h>

// ---------------------------------------------------------------------------
// MHA forward for MI455X (gfx1250, wave32, WMMA).
// d_out = [out (4*1024*1024 f32)] ++ [attn (4*16*1024*1024 f32)]
// ---------------------------------------------------------------------------

typedef __bf16 bf16;
typedef __attribute__((ext_vector_type(16))) __bf16 v16bf;
typedef __attribute__((ext_vector_type(8)))  float  v8f;

#define D_MODEL   1024
#define NUM_HEADS 16
#define D_K       64
#define BATCH     4
#define SEQ       1024
#define M_TOT     (BATCH * SEQ)

// ---- WMMA helpers ----------------------------------------------------------
// 16-bit A/B fragment layout (ISA 7.12.2): lane&15 selects the row (M for A,
// N for B); lane>>4 selects K-halves {0..7,16..23} vs {8..15,24..31}.
__device__ __forceinline__ v16bf frag_from_f32(const float* __restrict__ base,
                                               int ld, int lane) {
  const int r  = lane & 15;
  const int kh = (lane >> 4) << 3;
  const float* p = base + r * ld + kh;
  v16bf f;
#pragma unroll
  for (int i = 0; i < 8; ++i) {
    f[i]     = (bf16)p[i];
    f[8 + i] = (bf16)p[16 + i];
  }
  return f;
}

__device__ __forceinline__ v16bf frag_from_bf16(const bf16* __restrict__ base,
                                                int ld, int lane) {
  const int r  = lane & 15;
  const int kh = (lane >> 4) << 3;
  const bf16* p = base + r * ld + kh;
  v16bf f;
#pragma unroll
  for (int i = 0; i < 8; ++i) {
    f[i]     = p[i];
    f[8 + i] = p[16 + i];
  }
  return f;
}

__device__ __forceinline__ v8f wmma_bf16(v16bf a, v16bf b, v8f c) {
  // (neg_a, A, neg_b, B, c_mod, C, reuse_a, reuse_b)
  return __builtin_amdgcn_wmma_f32_16x16x32_bf16(false, a, false, b,
                                                 (short)0, c, false, false);
}

// ---- Kernel 1: projection GEMM  Y = X @ W^T + bias, head-split bf16 store --
// X: (4096,1024) f32 row-major.  W: (1024,1024) f32 [out,in] row-major, which
// is exactly the B-fragment layout (row = N, contiguous K) for X @ W^T.
// transposed==0: Y[b][h][s][d]   (q, k)
// transposed==1: Y[b][h][d][s]   (v stored pre-transposed for the PV GEMM)
__global__ void proj_kernel(const float* __restrict__ X,
                            const float* __restrict__ W,
                            const float* __restrict__ bias,
                            bf16* __restrict__ Y, int transposed) {
  const int lane = threadIdx.x & 31;
  const int wave = threadIdx.x >> 5;
  const int m0 = blockIdx.x * 16;
  const int n0 = (blockIdx.y * 8 + wave) * 16;

  const float* arow = X + (long)m0 * D_MODEL;
  const float* brow = W + (long)n0 * D_MODEL;
  v8f acc = {};
#pragma unroll 4
  for (int kk = 0; kk < D_MODEL; kk += 32) {
    v16bf a = frag_from_f32(arow + kk, D_MODEL, lane);
    v16bf b = frag_from_f32(brow + kk, D_MODEL, lane);
    acc = wmma_bf16(a, b, acc);
  }

  const int n = n0 + (lane & 15);
  const int h = n >> 6, d = n & 63;
  const float bn = bias[n];
  const int mbase = m0 + ((lane >> 4) << 3);
#pragma unroll
  for (int i = 0; i < 8; ++i) {
    const int m = mbase + i;
    const int b_ = m >> 10, s = m & 1023;
    const float v = acc[i] + bn;
    const long idx = transposed
        ? ((((long)b_ * NUM_HEADS + h) * D_K + d) * SEQ + s)
        : ((((long)b_ * NUM_HEADS + h) * SEQ + s) * D_K + d);
    Y[idx] = (bf16)v;
  }
}

// ---- Kernel 2: fused scores + softmax + attn-write + PV GEMM ---------------
// One block (8 waves) per (b, h, 16-query tile).  Scores live in 64 KB LDS
// (320 KB/WGP available), softmax is pure wave32 shuffle, attn is written
// exactly once (non-temporal: don't evict K/V head tiles from L2), and the
// probability tile feeds the PV WMMA straight from LDS.
__global__ void attn_fused_kernel(const bf16* __restrict__ qh,
                                  const bf16* __restrict__ kh,
                                  const bf16* __restrict__ vt,
                                  float* __restrict__ attn_out,
                                  bf16* __restrict__ ctx) {
  __shared__ float sc[16][SEQ];  // 64 KB
  const int lane = threadIdx.x & 31;
  const int wave = threadIdx.x >> 5;
  int t = blockIdx.x;
  const int qT = t & 63;  t >>= 6;
  const int h  = t & 15;  t >>= 4;
  const int b  = t;
  const long headOff = ((long)b * NUM_HEADS + h) * (long)SEQ * D_K;
  const bf16* qrow = qh + headOff + (long)qT * 16 * D_K;

  // Phase 1: S = (Q K^T) / 8 into LDS.  8 waves x 8 key tiles each.
  const v16bf aq0 = frag_from_bf16(qrow + 0,  D_K, lane);
  const v16bf aq1 = frag_from_bf16(qrow + 32, D_K, lane);
  for (int kT = wave; kT < SEQ / 16; kT += 8) {
    const bf16* krow = kh + headOff + (long)kT * 16 * D_K;
    v8f acc = {};
    acc = wmma_bf16(aq0, frag_from_bf16(krow + 0,  D_K, lane), acc);
    acc = wmma_bf16(aq1, frag_from_bf16(krow + 32, D_K, lane), acc);
    const int kcol  = kT * 16 + (lane & 15);
    const int rbase = (lane >> 4) << 3;
#pragma unroll
    for (int i = 0; i < 8; ++i) sc[rbase + i][kcol] = acc[i] * 0.125f;
  }
  __syncthreads();

  // Phase 2: row softmax (wave32 shuffle reduction), single NT write of attn.
  float* attnRow = attn_out + (((long)b * NUM_HEADS + h) * SEQ + qT * 16) * (long)SEQ;
  for (int r = wave; r < 16; r += 8) {
    float vals[32];
    float mx = -3.4e38f;
#pragma unroll
    for (int j = 0; j < 32; ++j) {
      vals[j] = sc[r][lane + j * 32];
      mx = fmaxf(mx, vals[j]);
    }
#pragma unroll
    for (int off = 16; off; off >>= 1) mx = fmaxf(mx, __shfl_xor(mx, off, 32));
    float sum = 0.f;
#pragma unroll
    for (int j = 0; j < 32; ++j) { vals[j] = __expf(vals[j] - mx); sum += vals[j]; }
#pragma unroll
    for (int off = 16; off; off >>= 1) sum += __shfl_xor(sum, off, 32);
    const float inv = 1.0f / sum;
#pragma unroll
    for (int j = 0; j < 32; ++j) {
      const float pv = vals[j] * inv;
      sc[r][lane + j * 32] = pv;
      __builtin_nontemporal_store(pv, &attnRow[(long)r * SEQ + lane + j * 32]);
    }
  }
  __syncthreads();

  // Phase 3: ctx(16 x 64) = P(16 x 1024) @ V(1024 x 64).  vt is (b,h,d,s), so
  // row d with contiguous s matches the B-fragment layout directly.
  // Wave-uniform branch: EXEC is all-ones inside (WMMA requirement).
  if (wave < 4) {
    const bf16* brow = vt + (((long)b * NUM_HEADS + h) * D_K + wave * 16) * (long)SEQ;
    v8f acc = {};
    for (int kk = 0; kk < SEQ; kk += 32) {
      v16bf a  = frag_from_f32(&sc[0][0] + kk, SEQ, lane);
      v16bf bm = frag_from_bf16(brow + kk, SEQ, lane);
      acc = wmma_bf16(a, bm, acc);
    }
    const int d     = wave * 16 + (lane & 15);
    const int qbase = qT * 16 + ((lane >> 4) << 3);
#pragma unroll
    for (int i = 0; i < 8; ++i) {
      // ctx stored directly as merged-head (B,S,D) row-major
      ctx[(((long)b * SEQ + qbase + i) * D_MODEL) + h * D_K + d] = (bf16)acc[i];
    }
  }
}

// ---- Kernel 3: output projection  out = ctx @ wo^T + bo (fp32 out) ---------
__global__ void outproj_kernel(const bf16* __restrict__ X,
                               const float* __restrict__ W,
                               const float* __restrict__ bias,
                               float* __restrict__ out) {
  const int lane = threadIdx.x & 31;
  const int wave = threadIdx.x >> 5;
  const int m0 = blockIdx.x * 16;
  const int n0 = (blockIdx.y * 8 + wave) * 16;

  const bf16*  arow = X + (long)m0 * D_MODEL;
  const float* brow = W + (long)n0 * D_MODEL;
  v8f acc = {};
#pragma unroll 4
  for (int kk = 0; kk < D_MODEL; kk += 32) {
    v16bf a  = frag_from_bf16(arow + kk, D_MODEL, lane);
    v16bf bm = frag_from_f32(brow + kk, D_MODEL, lane);
    acc = wmma_bf16(a, bm, acc);
  }

  const int n = n0 + (lane & 15);
  const float bn = bias[n];
  const int mbase = m0 + ((lane >> 4) << 3);
#pragma unroll
  for (int i = 0; i < 8; ++i)
    out[(long)(mbase + i) * D_MODEL + n] = acc[i] + bn;
}

// ---------------------------------------------------------------------------
extern "C" void kernel_launch(void* const* d_in, const int* in_sizes, int n_in,
                              void* d_out, int out_size, void* d_ws, size_t ws_size,
                              hipStream_t stream) {
  (void)in_sizes; (void)n_in; (void)out_size; (void)ws_size;

  const float* Q  = (const float*)d_in[0];
  const float* K  = (const float*)d_in[1];
  const float* V  = (const float*)d_in[2];
  const float* wq = (const float*)d_in[3];
  const float* bq = (const float*)d_in[4];
  const float* wk = (const float*)d_in[5];
  const float* bk = (const float*)d_in[6];
  const float* wv = (const float*)d_in[7];
  const float* bv = (const float*)d_in[8];
  const float* wo = (const float*)d_in[9];
  const float* bo = (const float*)d_in[10];

  float* out  = (float*)d_out;
  float* attn = out + (long)BATCH * SEQ * D_MODEL;

  const size_t headElems = (size_t)BATCH * NUM_HEADS * SEQ * D_K;  // 4M
  bf16* qh  = (bf16*)d_ws;          //  8 MB
  bf16* kh  = qh + headElems;       //  8 MB
  bf16* vt  = kh + headElems;       //  8 MB (stored (b,h,d,s))
  bf16* ctx = vt + headElems;       //  8 MB   -> 32 MB total workspace

  const dim3 blk(256);                               // 8 wave32 waves
  const dim3 gProj(M_TOT / 16, D_MODEL / 128);       // 256 x 8 blocks

  proj_kernel<<<gProj, blk, 0, stream>>>(Q, wq, bq, qh, 0);
  proj_kernel<<<gProj, blk, 0, stream>>>(K, wk, bk, kh, 0);
  proj_kernel<<<gProj, blk, 0, stream>>>(V, wv, bv, vt, 1);

  attn_fused_kernel<<<BATCH * NUM_HEADS * (SEQ / 16), blk, 0, stream>>>(
      qh, kh, vt, attn, ctx);

  outproj_kernel<<<gProj, blk, 0, stream>>>(ctx, wo, bo, out);
}
